// AttnBlock1D_13125420057120
// MI455X (gfx1250) — matrix operand order repaired and checked by reading, not compiled, not run on hardware
//
#include <hip/hip_runtime.h>

#define C_ 256
#define L_ 4096
#define B_ 4
#define EPS_ 1e-5f

typedef __attribute__((ext_vector_type(16))) _Float16 v16h;
typedef __attribute__((ext_vector_type(8)))  _Float16 v8h;
typedef __attribute__((ext_vector_type(8)))  float    v8f;

__device__ __forceinline__ v16h pack16(v8h lo, v8h hi) {
  v16h a;
#pragma unroll
  for (int e = 0; e < 8; ++e) { a[e] = lo[e]; a[e + 8] = hi[e]; }
  return a;
}

// A operand (16x32 f16). Per-lane: row M = lane&15 (caller bakes into rowp),
// elems 0..15 <- K = hi*16+0..15 (contiguous, matches B)
__device__ __forceinline__ v16h load_a16(const _Float16* rowp, int kb, int hi) {
  v8h lo = *(const v8h*)(rowp + kb * 32 + hi * 16);
  v8h hh = *(const v8h*)(rowp + kb * 32 + hi * 16 + 8);
  return pack16(lo, hh);
}

// B operand (32x16 f16). Per-lane: col N = lane&15 (baked into p),
// 16 contiguous K values starting at K = hi*16.
__device__ __forceinline__ v16h load_b16(const _Float16* p) {
  v8h lo = *(const v8h*)(p);
  v8h hh = *(const v8h*)(p + 8);
  return pack16(lo, hh);
}

__device__ __forceinline__ v8f wmma16(v16h a, v16h b, v8f c) {
  return __builtin_amdgcn_wmma_f32_16x16x32_f16(
      /*neg_a=*/false, a, /*neg_b=*/false, b,
      /*c_mod=*/(short)0, c, /*reuse_a=*/false, /*reuse_b=*/false);
}

// ---------------------------------------------------------------------------
// Kernel 1: per-channel BatchNorm statistics -> (scale, shift) per channel
// ---------------------------------------------------------------------------
__global__ void __launch_bounds__(256) bn_stats_kernel(
    const float* __restrict__ x, const float* __restrict__ gamma,
    const float* __restrict__ beta, float2* __restrict__ ss) {
  const int c = blockIdx.x, tid = threadIdx.x;
  float sum = 0.f, sq = 0.f;
  for (int b = 0; b < B_; ++b) {
    const float* xc = x + ((size_t)b * C_ + c) * L_;
    for (int l = tid; l < L_; l += 256) { float v = xc[l]; sum += v; sq += v * v; }
  }
  __shared__ float s1[256], s2[256];
  s1[tid] = sum; s2[tid] = sq; __syncthreads();
  for (int off = 128; off > 0; off >>= 1) {
    if (tid < off) { s1[tid] += s1[tid + off]; s2[tid] += s2[tid + off]; }
    __syncthreads();
  }
  if (tid == 0) {
    const float inv = 1.f / (float)(B_ * L_);
    float mean = s1[0] * inv;
    float var  = s2[0] * inv - mean * mean;   // biased, like torch BN
    float sc   = gamma[c] * rsqrtf(var + EPS_);
    ss[c] = make_float2(sc, beta[c] - mean * sc);
  }
}

// ---------------------------------------------------------------------------
// Kernel 2: convert the four CxC weight matrices to f16 (row-major [o][c])
// ---------------------------------------------------------------------------
__global__ void __launch_bounds__(256) cvt_w_kernel(
    const float* __restrict__ wq, const float* __restrict__ wk,
    const float* __restrict__ wv, const float* __restrict__ wp,
    _Float16* __restrict__ w16) {
  int i = blockIdx.x * 256 + threadIdx.x;     // 65536 elems per matrix
  w16[i]             = (_Float16)wq[i];
  w16[65536 + i]     = (_Float16)wk[i];
  w16[2 * 65536 + i] = (_Float16)wv[i];
  w16[3 * 65536 + i] = (_Float16)wp[i];
}

// ---------------------------------------------------------------------------
// Kernel 3: fused BN-apply + QKV projection via WMMA.
// One wave owns a 16-wide l-tile; B operand (normalized x, C x 16) is built
// once, reused across 3 weights x 16 o-tiles. A tiles preloaded as an array;
// two independent accumulator chains to keep the XDL pipe busy.
// ---------------------------------------------------------------------------
__global__ void __launch_bounds__(256, 1) attn_qkv_kernel(
    const float* __restrict__ x, const float2* __restrict__ ss,
    const _Float16* __restrict__ w16,
    const float* __restrict__ bq, const float* __restrict__ bk,
    const float* __restrict__ bv,
    _Float16* __restrict__ qT, _Float16* __restrict__ kT,
    _Float16* __restrict__ vv) {
  __shared__ float2 ssh[C_];
  __shared__ float  bsh[3][C_];
  for (int i = threadIdx.x; i < C_; i += blockDim.x) {
    ssh[i] = ss[i]; bsh[0][i] = bq[i]; bsh[1][i] = bk[i]; bsh[2][i] = bv[i];
  }
  __syncthreads();

  const int lane = threadIdx.x & 31, wave = threadIdx.x >> 5;
  const int gw = blockIdx.x * 8 + wave;       // 0..1023
  const int b = gw >> 8, lt = gw & 255;
  const int col = lane & 15, hi = lane >> 4;
  const int pos = lt * 16 + col;

  // B-operand tiles of h = x*scale+shift : bh[kb] covers K=c in [kb*32, kb*32+32)
  v16h bh[8];
  const float* xb = x + (size_t)b * C_ * L_;
#pragma unroll
  for (int kb = 0; kb < 8; ++kb) {
#pragma unroll
    for (int e = 0; e < 16; ++e) {
      int c = kb * 32 + hi * 16 + e;
      float2 s = ssh[c];
      bh[kb][e] = (_Float16)(xb[(size_t)c * L_ + pos] * s.x + s.y);
    }
  }

  for (int m = 0; m < 3; ++m) {
    const _Float16* wmat = w16 + (size_t)m * C_ * C_;
    const float qs = (m == 0) ? 0.0625f : 1.0f;   // fold 1/sqrt(256) into q
#pragma unroll 1
    for (int ot = 0; ot < 16; ++ot) {
      const _Float16* wrow = wmat + (size_t)(ot * 16 + col) * C_;  // A row M=lane&15
      v16h aw[8];
#pragma unroll
      for (int kb = 0; kb < 8; ++kb) aw[kb] = load_a16(wrow, kb, hi);
      v8f acc0 = {}, acc1 = {};
#pragma unroll
      for (int kb = 0; kb < 8; kb += 2) {
        acc0 = wmma16(aw[kb],     bh[kb],     acc0);
        acc1 = wmma16(aw[kb + 1], bh[kb + 1], acc1);
      }
      if (m == 2) {          // v -> [B][C][L] (j contiguous), row-coalesced b16 stores
#pragma unroll
        for (int r = 0; r < 8; ++r) {
          int o = ot * 16 + hi * 8 + r;
          vv[((size_t)b * C_ + o) * L_ + pos] =
              (_Float16)(acc0[r] + acc1[r] + bsh[2][o]);
        }
      } else {               // q/k -> [B][L][C] (c contiguous), one b128 store
        v8h t;
#pragma unroll
        for (int r = 0; r < 8; ++r) {
          int o = ot * 16 + hi * 8 + r;
          t[r] = (_Float16)((acc0[r] + acc1[r] + bsh[m][o]) * qs);
        }
        _Float16* dst = (m == 0 ? qT : kT) +
                        ((size_t)(b * L_) + pos) * C_ + ot * 16 + hi * 8;
        *(v8h*)dst = t;
      }
    }
  }
}

// ---------------------------------------------------------------------------
// Kernel 4: flash attention. One wave per (b, 16-row i-tile).
// S-phase: 4 independent WMMA chains, loads batched per 2-kb step.
// PV-phase: 16 independent chains, loads grouped by 4.
// ---------------------------------------------------------------------------
__global__ void __launch_bounds__(256, 1) attn_core_kernel(
    const _Float16* __restrict__ qT, const _Float16* __restrict__ kT,
    const _Float16* __restrict__ vv, _Float16* __restrict__ O16) {
  __shared__ __align__(16) _Float16 plds[8][16 * 32];   // 1KB per wave
  const int lane = threadIdx.x & 31, wave = threadIdx.x >> 5;
  const int gw = blockIdx.x * 8 + wave;
  const int b = gw >> 8, it = gw & 255;
  const int col = lane & 15, hi = lane >> 4;
  _Float16* pbuf = plds[wave];

  // Q rows as A-operand tiles (K = c, 8 blocks of 32)
  v16h aq[8];
  {
    const _Float16* qrow = qT + ((size_t)(b * L_) + it * 16 + col) * C_;
#pragma unroll
    for (int kb = 0; kb < 8; ++kb) aq[kb] = load_a16(qrow, kb, hi);
  }

  v8f oacc[16];
#pragma unroll
  for (int nt = 0; nt < 16; ++nt) oacc[nt] = (v8f){};
  float mrow[8], lrow[8];
#pragma unroll
  for (int r = 0; r < 8; ++r) { mrow[r] = -1e30f; lrow[r] = 0.f; }

  const _Float16* kTb = kT + (size_t)(b * L_) * C_;
  const _Float16* vb  = vv + (size_t)b * C_ * L_;

  for (int jb = 0; jb < L_ / 32; ++jb) {
    const int j0 = jb * 32;
    // ---- S = (Q^T K) * scale : 4 independent accumulator chains ---------
    v8f s0a = (v8f){}, s0b = (v8f){}, s1a = (v8f){}, s1b = (v8f){};
    {
      const _Float16* k0 = kTb + (size_t)(j0 + col) * C_ + hi * 16;
      const _Float16* k1 = kTb + (size_t)(j0 + 16 + col) * C_ + hi * 16;
#pragma unroll
      for (int kb = 0; kb < 8; kb += 2) {
        v16h b00 = load_b16(k0 + kb * 32);
        v16h b10 = load_b16(k1 + kb * 32);
        v16h b01 = load_b16(k0 + kb * 32 + 32);
        v16h b11 = load_b16(k1 + kb * 32 + 32);
        s0a = wmma16(aq[kb],     b00, s0a);
        s1a = wmma16(aq[kb],     b10, s1a);
        s0b = wmma16(aq[kb + 1], b01, s0b);
        s1b = wmma16(aq[kb + 1], b11, s1b);
      }
    }
    if (jb + 1 < L_ / 32)   // global_prefetch of next K block
      __builtin_prefetch(kTb + (size_t)(j0 + 32 + col) * C_, 0, 1);

    // ---- online softmax: row stats across the 16 lanes of each half -----
    float pr0[8], pr1[8], asc[8];
#pragma unroll
    for (int r = 0; r < 8; ++r) {
      float v0 = s0a[r] + s0b[r];
      float v1 = s1a[r] + s1b[r];
      float mx = fmaxf(v0, v1);
      mx = fmaxf(mx, __shfl_xor(mx, 1, 32));
      mx = fmaxf(mx, __shfl_xor(mx, 2, 32));
      mx = fmaxf(mx, __shfl_xor(mx, 4, 32));
      mx = fmaxf(mx, __shfl_xor(mx, 8, 32));
      float mnew = fmaxf(mrow[r], mx);
      asc[r] = __expf(mrow[r] - mnew);
      mrow[r] = mnew;
      pr0[r] = __expf(v0 - mnew);
      pr1[r] = __expf(v1 - mnew);
      float rs = pr0[r] + pr1[r];
      rs += __shfl_xor(rs, 1, 32);
      rs += __shfl_xor(rs, 2, 32);
      rs += __shfl_xor(rs, 4, 32);
      rs += __shfl_xor(rs, 8, 32);
      lrow[r] = lrow[r] * asc[r] + rs;
    }
#pragma unroll
    for (int nt = 0; nt < 16; ++nt)
#pragma unroll
      for (int r = 0; r < 8; ++r) oacc[nt][r] *= asc[r];

    // ---- transpose P (D layout, lane=col) -> A layout (lane=row) via LDS -
#pragma unroll
    for (int r = 0; r < 8; ++r) {
      int row = hi * 8 + r;
      pbuf[row * 32 + col]      = (_Float16)pr0[r];
      pbuf[row * 32 + 16 + col] = (_Float16)pr1[r];
    }
    // per-wave LDS is in-order; fence compiler + wait DS stores before reload
    asm volatile("s_wait_dscnt 0" ::: "memory");
    v16h aP;
    {
      const _Float16* prow = pbuf + col * 32;   // row M = lane&15
      v8h lo = *(const v8h*)(prow + hi * 16);
      v8h hh = *(const v8h*)(prow + hi * 16 + 8);
      aP = pack16(lo, hh);
    }
    asm volatile("" ::: "memory");

    // ---- O += P * V^T : 16 independent chains, loads grouped by 4 -------
#pragma unroll
    for (int nt = 0; nt < 16; nt += 4) {
      v16h vb0 = load_b16(vb + (size_t)((nt + 0) * 16 + col) * L_ + j0 + hi * 16);
      v16h vb1 = load_b16(vb + (size_t)((nt + 1) * 16 + col) * L_ + j0 + hi * 16);
      v16h vb2 = load_b16(vb + (size_t)((nt + 2) * 16 + col) * L_ + j0 + hi * 16);
      v16h vb3 = load_b16(vb + (size_t)((nt + 3) * 16 + col) * L_ + j0 + hi * 16);
      oacc[nt + 0] = wmma16(aP, vb0, oacc[nt + 0]);
      oacc[nt + 1] = wmma16(aP, vb1, oacc[nt + 1]);
      oacc[nt + 2] = wmma16(aP, vb2, oacc[nt + 2]);
      oacc[nt + 3] = wmma16(aP, vb3, oacc[nt + 3]);
    }
  }

  // ---- finalize: divide by row sums, store O as [B][L][C] f16 -----------
  float rl[8];
#pragma unroll
  for (int r = 0; r < 8; ++r) rl[r] = 1.f / lrow[r];
  _Float16* ob = O16 + ((size_t)(b * L_) + it * 16) * C_;
#pragma unroll
  for (int nt = 0; nt < 16; ++nt) {
    int c = nt * 16 + col;
#pragma unroll
    for (int r = 0; r < 8; ++r) {
      int irow = hi * 8 + r;
      ob[(size_t)irow * C_ + c] = (_Float16)(oacc[nt][r] * rl[r]);
    }
  }
}

// ---------------------------------------------------------------------------
// Kernel 5: projection + residual. out[b][o][i] = x + bp[o] + sum_c wp[o][c]*O[i][c]
// ---------------------------------------------------------------------------
__global__ void __launch_bounds__(256, 1) attn_proj_kernel(
    const float* __restrict__ x, const _Float16* __restrict__ O16,
    const _Float16* __restrict__ wp16, const float* __restrict__ bp,
    float* __restrict__ out) {
  __shared__ float bps[C_];
  for (int i = threadIdx.x; i < C_; i += blockDim.x) bps[i] = bp[i];
  __syncthreads();
  const int lane = threadIdx.x & 31, wave = threadIdx.x >> 5;
  const int gw = blockIdx.x * 8 + wave;
  const int b = gw >> 8, it = gw & 255;
  const int col = lane & 15, hi = lane >> 4;
  const int ipos = it * 16 + col;

  v16h bo[8];
  const _Float16* orow = O16 + ((size_t)(b * L_) + ipos) * C_ + hi * 16;
#pragma unroll
  for (int kb = 0; kb < 8; ++kb) bo[kb] = load_b16(orow + kb * 32);

#pragma unroll 1
  for (int ot = 0; ot < 16; ++ot) {
    const _Float16* wrow = wp16 + (size_t)(ot * 16 + col) * C_;
    v16h aw[8];
#pragma unroll
    for (int kb = 0; kb < 8; ++kb) aw[kb] = load_a16(wrow, kb, hi);
    v8f acc0 = {}, acc1 = {};
#pragma unroll
    for (int kb = 0; kb < 8; kb += 2) {
      acc0 = wmma16(aw[kb],     bo[kb],     acc0);
      acc1 = wmma16(aw[kb + 1], bo[kb + 1], acc1);
    }
#pragma unroll
    for (int r = 0; r < 8; ++r) {
      int o = ot * 16 + hi * 8 + r;
      size_t idx = ((size_t)b * C_ + o) * L_ + ipos;
      out[idx] = x[idx] + bps[o] + acc0[r] + acc1[r];
    }
  }
}

// ---------------------------------------------------------------------------
extern "C" void kernel_launch(void* const* d_in, const int* in_sizes, int n_in,
                              void* d_out, int out_size, void* d_ws, size_t ws_size,
                              hipStream_t stream) {
  (void)in_sizes; (void)n_in; (void)out_size; (void)ws_size;
  const float* x     = (const float*)d_in[0];
  const float* gamma = (const float*)d_in[1];
  const float* beta  = (const float*)d_in[2];
  const float* wq    = (const float*)d_in[3];
  const float* bq    = (const float*)d_in[4];
  const float* wk    = (const float*)d_in[5];
  const float* bk    = (const float*)d_in[6];
  const float* wv    = (const float*)d_in[7];
  const float* bv    = (const float*)d_in[8];
  const float* wp    = (const float*)d_in[9];
  const float* bp    = (const float*)d_in[10];
  float* out = (float*)d_out;

  // workspace layout (~32.5 MiB): ss | w16(q,k,v,p) | qT | kT | v | O
  char* wsb = (char*)d_ws;
  float2*   ss  = (float2*)wsb;
  _Float16* w16 = (_Float16*)(wsb + 2048);
  size_t off = 2048 + (size_t)4 * 65536 * 2;
  const size_t tile_bytes = (size_t)B_ * L_ * C_ * 2;
  _Float16* qT  = (_Float16*)(wsb + off); off += tile_bytes;
  _Float16* kT  = (_Float16*)(wsb + off); off += tile_bytes;
  _Float16* vv  = (_Float16*)(wsb + off); off += tile_bytes;
  _Float16* O16 = (_Float16*)(wsb + off);

  const int nblk = B_ * L_ / (16 * 8);  // 128 blocks x 8 waves = 1024 wave-tiles

  hipLaunchKernelGGL(bn_stats_kernel, dim3(C_), dim3(256), 0, stream,
                     x, gamma, beta, ss);
  hipLaunchKernelGGL(cvt_w_kernel, dim3(256), dim3(256), 0, stream,
                     wq, wk, wv, wp, w16);
  hipLaunchKernelGGL(attn_qkv_kernel, dim3(nblk), dim3(256), 0, stream,
                     x, ss, w16, bq, bk, bv, qT, kT, vv);
  hipLaunchKernelGGL(attn_core_kernel, dim3(nblk), dim3(256), 0, stream,
                     qT, kT, vv, O16);
  hipLaunchKernelGGL(attn_proj_kernel, dim3(nblk), dim3(256), 0, stream,
                     x, O16, w16 + (size_t)3 * 65536, bp, out);
}
